// MultiModalGNN_5325759447716
// MI455X (gfx1250) — matrix operand order repaired and checked by reading, not compiled, not run on hardware
//
#include <hip/hip_runtime.h>
#include <hip/hip_bf16.h>

// ---------------------------------------------------------------------------
// Problem constants (match reference)
// ---------------------------------------------------------------------------
#define NU 80000
#define NP 40000
#define NE 160000
#define DM 256
#define NH 4
#define HD 64
#define INV_SQRT_D 0.125f  // 1/sqrt(64)

typedef __bf16 v16bf __attribute__((ext_vector_type(16)));
typedef float  v8f   __attribute__((ext_vector_type(8)));

union BFrag {
    v16bf v;
    uint4 q[2];
    unsigned short u[16];
};

// ---------------------------------------------------------------------------
// Helpers
// ---------------------------------------------------------------------------
__device__ __forceinline__ unsigned short f2bf(float f) {
    unsigned u = __float_as_uint(f);
    unsigned r = (u + 0x7FFFu + ((u >> 16) & 1u)) >> 16;  // RNE
    return (unsigned short)r;
}
__device__ __forceinline__ unsigned pack2bf(float a, float b) {
    return (unsigned)f2bf(a) | ((unsigned)f2bf(b) << 16);
}
__device__ __forceinline__ float bf2f(unsigned short h) {
    return __uint_as_float(((unsigned)h) << 16);
}
__device__ __forceinline__ float gelu_exact(float x) {
    return 0.5f * x * (1.0f + erff(x * 0.7071067811865476f));
}
// order-preserving float -> uint encoding for atomicMax-based segment max
__device__ __forceinline__ unsigned encU(float f) {
    unsigned u = __float_as_uint(f);
    return (u & 0x80000000u) ? ~u : (u | 0x80000000u);
}
__device__ __forceinline__ float decU(unsigned u) {
    return (u & 0x80000000u) ? __uint_as_float(u & 0x7FFFFFFFu)
                             : __uint_as_float(~u);
}
// dot of two packed bf16 pairs held in a uint
__device__ __forceinline__ float pairdot(unsigned a, unsigned b) {
    float a0 = __uint_as_float(a << 16);
    float a1 = __uint_as_float(a & 0xFFFF0000u);
    float b0 = __uint_as_float(b << 16);
    float b1 = __uint_as_float(b & 0xFFFF0000u);
    return a0 * b0 + a1 * b1;
}

// ---------------------------------------------------------------------------
// WMMA GEMM: Out[M,N] = op(A)[M,K] * B[K,N] + bias (+ skip-blend)
// AMODE: 0 = A is f32, 1 = A is f32 with exact GELU applied, 2 = A is bf16
// OMODE: 0 = write bf16, 1 = write f32, 2 = f32 blend: beta*v+(1-beta)*Xold
// Block: 128 threads (4 waves). Block tile 64x64, K-step 32.
// Requires M%64==0, N%64==0, K%32==0 (true for all uses here).
// LDS rows are 40 halfs (80 B) so 16-byte vector casts stay aligned.
// ---------------------------------------------------------------------------
template <int AMODE, int OMODE>
__global__ __launch_bounds__(128) void gemm_wmma(
    const void* __restrict__ Aptr, int lda,
    const float* __restrict__ B, int ldb,
    const float* __restrict__ bias,
    void* __restrict__ Out, int ldo,
    const float* __restrict__ Xold, int ldx,
    const float* __restrict__ skipPtr,
    int M, int N, int K)
{
    __shared__ unsigned short sA[64][40];   // A tile: 64 rows x 32 k (+8 pad)
    __shared__ unsigned short sBT[64][40];  // B tile TRANSPOSED: 64 n x 32 k (+8 pad)

    const int tid  = threadIdx.x;
    const int wave = tid >> 5;
    const int lane = tid & 31;
    const int m0 = blockIdx.y * 64;
    const int n0 = blockIdx.x * 64;

    v8f acc[4];
#pragma unroll
    for (int c = 0; c < 4; ++c)
#pragma unroll
        for (int r = 0; r < 8; ++r) acc[c][r] = 0.0f;

    for (int k0 = 0; k0 < K; k0 += 32) {
        // --- stage A tile (64x32), 4 elems per thread per iter ---
#pragma unroll
        for (int i = 0; i < 4; ++i) {
            int idx = (i * 128 + tid) * 4;
            int r = idx >> 5, c = idx & 31;
            if (AMODE == 2) {
                const unsigned short* ap =
                    (const unsigned short*)Aptr + (size_t)(m0 + r) * lda + (k0 + c);
                *(uint2*)&sA[r][c] = *(const uint2*)ap;
            } else {
                const float* ap =
                    (const float*)Aptr + (size_t)(m0 + r) * lda + (k0 + c);
                float4 f = *(const float4*)ap;
                if (AMODE == 1) {
                    f.x = gelu_exact(f.x);
                    f.y = gelu_exact(f.y);
                    f.z = gelu_exact(f.z);
                    f.w = gelu_exact(f.w);
                }
                unsigned* dst = (unsigned*)&sA[r][c];
                dst[0] = pack2bf(f.x, f.y);
                dst[1] = pack2bf(f.z, f.w);
            }
        }
        // --- stage B tile (32x64) transposed into sBT[n][k] ---
#pragma unroll
        for (int i = 0; i < 4; ++i) {
            int idx = (i * 128 + tid) * 4;
            int r = idx >> 6, c = idx & 63;
            float4 f = *(const float4*)&B[(size_t)(k0 + r) * ldb + (n0 + c)];
            sBT[c + 0][r] = f2bf(f.x);
            sBT[c + 1][r] = f2bf(f.y);
            sBT[c + 2][r] = f2bf(f.z);
            sBT[c + 3][r] = f2bf(f.w);
        }
        __syncthreads();

        // --- A fragment: 16x32 bf16 (ISA layout), two b128 LDS loads ---
        BFrag a;
        const int ml = wave * 16 + (lane & 15);
        const int ah = (lane >> 4) ? 8 : 0;
        a.q[0] = *(const uint4*)&sA[ml][ah];       // K = ah..ah+7
        a.q[1] = *(const uint4*)&sA[ml][16 + ah];  // K = 16+ah..16+ah+7

        // --- 4 column tiles of B (32x16 each), two b128 loads each ---
        const int nl  = lane & 15;
        const int kbB = (lane >> 4) * 8;
#pragma unroll
        for (int c = 0; c < 4; ++c) {
            BFrag b;
            const int cn = c * 16 + nl;
            b.q[0] = *(const uint4*)&sBT[cn][kbB];      // K = kbB..kbB+7
            b.q[1] = *(const uint4*)&sBT[cn][16 + kbB];  // K = 16+kbB..16+kbB+7
            acc[c] = __builtin_amdgcn_wmma_f32_16x16x32_bf16(
                false, a.v, false, b.v, (short)0, acc[c], false, false);
        }
        __syncthreads();
    }

    // --- epilogue ---
    float beta = 0.0f;
    if (OMODE == 2) {
        float sv = skipPtr[0];
        beta = 1.0f / (1.0f + expf(-sv));
    }
    const int rowBase = m0 + wave * 16 + ((lane >> 4) << 3);
    const int colBase = n0 + (lane & 15);
#pragma unroll
    for (int c = 0; c < 4; ++c) {
        int gn = colBase + c * 16;
        float bv = bias ? bias[gn] : 0.0f;
#pragma unroll
        for (int r = 0; r < 8; ++r) {
            int gm = rowBase + r;
            float v = acc[c][r] + bv;
            if (OMODE == 0) {
                ((unsigned short*)Out)[(size_t)gm * ldo + gn] = f2bf(v);
            } else if (OMODE == 1) {
                ((float*)Out)[(size_t)gm * ldo + gn] = v;
            } else {
                float old = Xold[(size_t)gm * ldx + gn];
                ((float*)Out)[(size_t)gm * ldo + gn] = beta * v + (1.0f - beta) * old;
            }
        }
    }
}

// ---------------------------------------------------------------------------
// Small utility kernels
// ---------------------------------------------------------------------------
__global__ void zero_f32(float* p, int n) {
    int g = blockIdx.x * blockDim.x + threadIdx.x;
    if (g < n) p[g] = 0.0f;
}
__global__ void init_u32(unsigned* p, unsigned v, int n) {
    int g = blockIdx.x * blockDim.x + threadIdx.x;
    if (g < n) p[g] = v;
}
__global__ void gather_rows(const float* __restrict__ table,
                            const int* __restrict__ ids,
                            float* __restrict__ out, int nrows) {
    int g = blockIdx.x * blockDim.x + threadIdx.x;
    if (g >= nrows * DM) return;
    int row = g >> 8, col = g & 255;
    out[g] = table[(size_t)ids[row] * DM + col];
}
__global__ void gelu_inplace(float* p, int n) {
    int g = blockIdx.x * blockDim.x + threadIdx.x;
    if (g < n) p[g] = gelu_exact(p[g]);
}

// ---------------------------------------------------------------------------
// Edge kernels (1 thread per (edge, head))
// ---------------------------------------------------------------------------
__global__ void edge_logits(const int* __restrict__ srcA,
                            const int* __restrict__ dstA,
                            const unsigned short* __restrict__ q,   // [n_dst,256] bf16
                            const unsigned short* __restrict__ kt,  // [n_src,256] bf16
                            const float* __restrict__ prel,         // [H]
                            float* __restrict__ logits, int nE) {
    int g = blockIdx.x * blockDim.x + threadIdx.x;
    if (g >= nE * NH) return;
    int e = g >> 2, h = g & 3;
    int s = srcA[e], d = dstA[e];
    const uint4* q4 = (const uint4*)(q + (size_t)d * DM + h * HD);
    const uint4* k4 = (const uint4*)(kt + (size_t)s * DM + h * HD);
    float acc = 0.0f;
#pragma unroll
    for (int i = 0; i < 8; ++i) {
        uint4 a = q4[i];
        uint4 b = k4[i];
        acc += pairdot(a.x, b.x) + pairdot(a.y, b.y) +
               pairdot(a.z, b.z) + pairdot(a.w, b.w);
    }
    logits[g] = acc * prel[h] * INV_SQRT_D;
}

__global__ void seg_max(const int* __restrict__ dstA,
                        const float* __restrict__ logits,
                        unsigned* __restrict__ menc, int nE) {
    int g = blockIdx.x * blockDim.x + threadIdx.x;
    if (g >= nE * NH) return;
    int e = g >> 2, h = g & 3;
    atomicMax(&menc[(size_t)dstA[e] * NH + h], encU(logits[g]));
}

__global__ void seg_expsum(const int* __restrict__ dstA,
                           const float* __restrict__ logits,
                           const unsigned* __restrict__ menc,
                           float* __restrict__ ssum, int nE) {
    int g = blockIdx.x * blockDim.x + threadIdx.x;
    if (g >= nE * NH) return;
    int e = g >> 2, h = g & 3;
    int idx = dstA[e] * NH + h;
    atomicAdd(&ssum[idx], expf(logits[g] - decU(menc[idx])));
}

__global__ void edge_agg(const int* __restrict__ srcA,
                         const int* __restrict__ dstA,
                         const float* __restrict__ logits,
                         const unsigned* __restrict__ menc,
                         const float* __restrict__ ssum,
                         const unsigned short* __restrict__ vt,  // [n_src,256] bf16
                         float* __restrict__ agg,                // [n_dst,256] f32
                         int nE) {
    int g = blockIdx.x * blockDim.x + threadIdx.x;
    if (g >= nE * NH) return;
    int e = g >> 2, h = g & 3;
    int s = srcA[e], d = dstA[e];
    int idx = d * NH + h;
    float a = expf(logits[g] - decU(menc[idx])) / (ssum[idx] + 1e-16f);
    const uint4* v4 = (const uint4*)(vt + (size_t)s * DM + h * HD);
    float* ap = agg + (size_t)d * DM + h * HD;
#pragma unroll
    for (int i = 0; i < 8; ++i) {
        uint4 w = v4[i];
        atomicAdd(&ap[i * 8 + 0], a * bf2f((unsigned short)(w.x & 0xFFFF)));
        atomicAdd(&ap[i * 8 + 1], a * bf2f((unsigned short)(w.x >> 16)));
        atomicAdd(&ap[i * 8 + 2], a * bf2f((unsigned short)(w.y & 0xFFFF)));
        atomicAdd(&ap[i * 8 + 3], a * bf2f((unsigned short)(w.y >> 16)));
        atomicAdd(&ap[i * 8 + 4], a * bf2f((unsigned short)(w.z & 0xFFFF)));
        atomicAdd(&ap[i * 8 + 5], a * bf2f((unsigned short)(w.z >> 16)));
        atomicAdd(&ap[i * 8 + 6], a * bf2f((unsigned short)(w.w & 0xFFFF)));
        atomicAdd(&ap[i * 8 + 7], a * bf2f((unsigned short)(w.w >> 16)));
    }
}

// ---------------------------------------------------------------------------
// Host orchestration
// ---------------------------------------------------------------------------
extern "C" void kernel_launch(void* const* d_in, const int* in_sizes, int n_in,
                              void* d_out, int out_size, void* d_ws, size_t ws_size,
                              hipStream_t stream) {
    const int*   user_ids = (const int*)d_in[0];
    const float* x_prod   = (const float*)d_in[1];
    const int*   e_src    = (const int*)d_in[2];
    const int*   e_dst    = (const int*)d_in[3];
    const float* emb      = (const float*)d_in[4];
    const float* Wk   = (const float*)d_in[5];
    const float* bk   = (const float*)d_in[6];
    const float* Wq   = (const float*)d_in[7];
    const float* bq   = (const float*)d_in[8];
    const float* Wv   = (const float*)d_in[9];
    const float* bv   = (const float*)d_in[10];
    const float* Wout = (const float*)d_in[11];
    const float* bout = (const float*)d_in[12];
    const float* skip = (const float*)d_in[13];
    const float* arel = (const float*)d_in[14];
    const float* mrel = (const float*)d_in[15];
    const float* prel = (const float*)d_in[16];
    float* out = (float*)d_out;

    // workspace carve-out
    char* ws = (char*)d_ws;
    auto carve = [&](size_t bytes) -> char* {
        char* p = ws;
        ws += (bytes + 255) & ~(size_t)255;
        return p;
    };
    const size_t Uf = (size_t)NU * DM * sizeof(float);
    const size_t Pf = (size_t)NP * DM * sizeof(float);
    const size_t Uh = (size_t)NU * DM * sizeof(unsigned short);
    const size_t Ph = (size_t)NP * DM * sizeof(unsigned short);

    float* xu_cur = (float*)carve(Uf);
    float* xp_cur = (float*)carve(Pf);
    float* xu_new = (float*)carve(Uf);
    float* xp_new = (float*)carve(Pf);
    float* agg_u  = (float*)carve(Uf);
    float* agg_p  = (float*)carve(Pf);
    unsigned short* q_u   = (unsigned short*)carve(Uh);
    unsigned short* kt_u  = (unsigned short*)carve(Uh);
    unsigned short* vt_u  = (unsigned short*)carve(Uh);
    unsigned short* tmp_u = (unsigned short*)carve(Uh);
    unsigned short* q_p   = (unsigned short*)carve(Ph);
    unsigned short* kt_p  = (unsigned short*)carve(Ph);
    unsigned short* vt_p  = (unsigned short*)carve(Ph);
    unsigned short* tmp_p = (unsigned short*)carve(Ph);
    float*    logits = (float*)carve((size_t)NE * NH * sizeof(float));
    unsigned* menc   = (unsigned*)carve((size_t)NU * NH * sizeof(unsigned));
    float*    ssum   = (float*)carve((size_t)NU * NH * sizeof(float));

    const dim3 blk128(128);
    const int EH = NE * NH;
    const int ebk = (EH + 255) / 256;

    auto gemmProj = [&](const float* A, const float* Bw, const float* bs,
                        unsigned short* O, int M) {
        dim3 g(DM / 64, M / 64);
        gemm_wmma<0, 0><<<g, blk128, 0, stream>>>(A, DM, Bw, DM, bs, O, DM,
                                                  nullptr, 0, nullptr, M, DM, DM);
    };
    auto gemmHead = [&](const unsigned short* A, const float* Bw,
                        unsigned short* O, int M) {
        dim3 g(1, M / 64);
        gemm_wmma<2, 0><<<g, blk128, 0, stream>>>(A, DM, Bw, HD, nullptr, O, DM,
                                                  nullptr, 0, nullptr, M, HD, HD);
    };
    auto gemmOut = [&](const float* A, const float* Bw, const float* bs,
                       const float* Xold, const float* skipP, float* O, int M) {
        dim3 g(DM / 64, M / 64);
        gemm_wmma<1, 2><<<g, blk128, 0, stream>>>(A, DM, Bw, DM, bs, O, DM,
                                                  Xold, DM, skipP, M, DM, DM);
    };
    auto relation = [&](const int* sA, const int* dA, const unsigned short* q,
                        const unsigned short* kt, const unsigned short* vt,
                        const float* prelP, int nDst, float* agg) {
        int nh = nDst * NH;
        init_u32<<<(nh + 255) / 256, 256, 0, stream>>>(menc, 0u, nh);
        zero_f32<<<(nh + 255) / 256, 256, 0, stream>>>(ssum, nh);
        zero_f32<<<(nDst * DM + 255) / 256, 256, 0, stream>>>(agg, nDst * DM);
        edge_logits<<<ebk, 256, 0, stream>>>(sA, dA, q, kt, prelP, logits, NE);
        seg_max<<<ebk, 256, 0, stream>>>(dA, logits, menc, NE);
        seg_expsum<<<ebk, 256, 0, stream>>>(dA, logits, menc, ssum, NE);
        edge_agg<<<ebk, 256, 0, stream>>>(sA, dA, logits, menc, ssum, vt, agg, NE);
    };

    // initial features
    gather_rows<<<(NU * DM + 255) / 256, 256, 0, stream>>>(emb, user_ids, xu_cur, NU);
    hipMemcpyAsync(xp_cur, x_prod, Pf, hipMemcpyDeviceToDevice, stream);

    float* curU = xu_cur;
    float* curP = xp_cur;
    const size_t WSZ = (size_t)DM * DM;  // 65536 per (l,t) weight block
    const size_t RSZ = (size_t)HD * HD;  // 4096 per (l,rel,h) relation block

    for (int l = 0; l < 2; ++l) {
        int lt0 = l * 2 + 0, lt1 = l * 2 + 1;

        // ---- user projections (type 0, src side of rel 0) ----
        gemmProj(curU, Wk + lt0 * WSZ, bk + lt0 * DM, tmp_u, NU);
        for (int h = 0; h < NH; ++h)
            gemmHead(tmp_u + h * HD, arel + ((size_t)lt0 * NH + h) * RSZ,
                     kt_u + h * HD, NU);
        gemmProj(curU, Wv + lt0 * WSZ, bv + lt0 * DM, tmp_u, NU);
        for (int h = 0; h < NH; ++h)
            gemmHead(tmp_u + h * HD, mrel + ((size_t)lt0 * NH + h) * RSZ,
                     vt_u + h * HD, NU);
        gemmProj(curU, Wq + lt0 * WSZ, bq + lt0 * DM, q_u, NU);

        // ---- product projections (type 1, src side of rel 1) ----
        gemmProj(curP, Wk + lt1 * WSZ, bk + lt1 * DM, tmp_p, NP);
        for (int h = 0; h < NH; ++h)
            gemmHead(tmp_p + h * HD, arel + ((size_t)lt1 * NH + h) * RSZ,
                     kt_p + h * HD, NP);
        gemmProj(curP, Wv + lt1 * WSZ, bv + lt1 * DM, tmp_p, NP);
        for (int h = 0; h < NH; ++h)
            gemmHead(tmp_p + h * HD, mrel + ((size_t)lt1 * NH + h) * RSZ,
                     vt_p + h * HD, NP);
        gemmProj(curP, Wq + lt1 * WSZ, bq + lt1 * DM, q_p, NP);

        float* outU = (l == 0) ? xu_new : out;
        float* outP = (l == 0) ? xp_new : (out + (size_t)NU * DM);

        // rel 0: user -> product (dst = product)
        relation(e_src, e_dst, q_p, kt_u, vt_u, prel + lt0 * NH, NP, agg_p);
        gemmOut(agg_p, Wout + lt1 * WSZ, bout + lt1 * DM, curP, skip + lt1, outP, NP);

        // rel 1: product -> user (dst = user)
        relation(e_dst, e_src, q_u, kt_p, vt_p, prel + lt1 * NH, NU, agg_u);
        gemmOut(agg_u, Wout + lt0 * WSZ, bout + lt0 * DM, curU, skip + lt0, outU, NU);

        if (l == 0) {
            gelu_inplace<<<(NU * DM + 255) / 256, 256, 0, stream>>>(xu_new, NU * DM);
            gelu_inplace<<<(NP * DM + 255) / 256, 256, 0, stream>>>(xp_new, NP * DM);
            curU = xu_new;
            curP = xp_new;
        }
    }
}
